// Wav2vecCriterionTorch_26044681683248
// MI455X (gfx1250) — matrix operand order repaired and checked by reading, not compiled, not run on hardware
//
#include <hip/hip_runtime.h>

// Problem constants (from reference): B=8, T=512, D=768, M=256, NUM_NEG=100
#define Bc 8
#define Tc 512
#define Dc 768
#define Mc 256
#define NNEG 100
#define ROWS (Bc * Mc)          // 2048 gathered rows
#define INV_TEMP 10.0f          // 1 / 0.1

typedef __attribute__((ext_vector_type(8)))  float  v8f;
typedef __attribute__((ext_vector_type(16))) __bf16 v16bf;

union FragBF16 { v16bf v; uint4 u[2]; };

__device__ __forceinline__ unsigned short f2bf(float f) {
    union { float f; unsigned int u; } c; c.f = f;
    unsigned int u = c.u;
    unsigned int r = u + 0x7FFFu + ((u >> 16) & 1u);   // round-to-nearest-even
    return (unsigned short)(r >> 16);
}

__device__ __forceinline__ void load_frag(FragBF16& f, const unsigned short* p0,
                                          const unsigned short* p1) {
    f.u[0] = *(const uint4*)p0;
    f.u[1] = *(const uint4*)p1;
}

// ---------------- K0: transpose+convert weight fp32[768,768] -> bf16 W^T ----
__global__ void k_wt(const float* __restrict__ W, unsigned short* __restrict__ Wt) {
    int i = blockIdx.x * blockDim.x + threadIdx.x;
    if (i < Dc * Dc) {
        int k = i / Dc, n = i % Dc;
        Wt[(size_t)n * Dc + k] = f2bf(W[(size_t)k * Dc + n]);
    }
}

// ---------------- K1: per-batch stable scan of boolean mask -> idx[b][m] ----
__global__ void k_mask_idx(const unsigned char* __restrict__ mask, int* __restrict__ idxbuf) {
    int b = threadIdx.x;
    if (b < Bc) {
        int cnt = 0;
        for (int t = 0; t < Tc && cnt < Mc; ++t)
            if (mask[b * Tc + t]) { idxbuf[b * Mc + cnt] = t; ++cnt; }
    }
}

// ---------------- K2: gather masked rows, convert to bf16 -------------------
__global__ void k_gather(const float* __restrict__ cnn, const float* __restrict__ quant,
                         const int* __restrict__ idxbuf,
                         unsigned short* __restrict__ Abf, unsigned short* __restrict__ Qbf) {
    int row = blockIdx.x;               // 0..2047
    int b = row >> 8;
    int t = idxbuf[row];
    const float* c = cnn   + ((size_t)b * Tc + t) * Dc;
    const float* q = quant + ((size_t)b * Tc + t) * Dc;
    for (int e = threadIdx.x; e < Dc; e += blockDim.x) {
        Abf[(size_t)row * Dc + e] = f2bf(c[e]);
        Qbf[(size_t)row * Dc + e] = f2bf(q[e]);
    }
}

// ---------------- K3: projection GEMM  C[2048,768] = A @ W + bias -----------
// A: bf16 row-major [2048,768]; Bt: bf16 [768,768] = W^T so the B-fragment
// (col n, consecutive K) is contiguous. One wave computes a 32x32 output tile
// (2x2 register blocking, 4 WMMAs per K-step of 32).
__global__ void k_gemm_proj(const unsigned short* __restrict__ A,
                            const unsigned short* __restrict__ Bt,
                            const float* __restrict__ bias,
                            float* __restrict__ C) {
    const int lane = threadIdx.x & 31;
    const int wave = threadIdx.x >> 5;
    const int tile = blockIdx.x * (blockDim.x >> 5) + wave;   // 64*24 tiles
    const int tr = tile / (Dc / 32);
    const int tc = tile % (Dc / 32);
    const int half = lane >> 4;   // K-half select (A: +8 of K, B: +8 of K)
    const int l16  = lane & 15;   // A: M row within tile; B: N col within tile
    const unsigned short* Ar0 = A  + (size_t)(tr * 32 + l16) * Dc + half * 8;
    const unsigned short* Ar1 = Ar0 + (size_t)16 * Dc;
    const unsigned short* Br0 = Bt + (size_t)(tc * 32 + l16) * Dc + half * 8;
    const unsigned short* Br1 = Br0 + (size_t)16 * Dc;
    const float bv0 = bias[tc * 32 + l16];
    const float bv1 = bias[tc * 32 + 16 + l16];
    v8f c00, c01, c10, c11;
#pragma unroll
    for (int v = 0; v < 8; ++v) { c00[v] = bv0; c01[v] = bv1; c10[v] = bv0; c11[v] = bv1; }
    for (int kk = 0; kk < Dc; kk += 32) {
        FragBF16 a0, a1, b0, b1;
        load_frag(a0, Ar0 + kk, Ar0 + kk + 16);   // K = kk+half*8..+7, +16..
        load_frag(a1, Ar1 + kk, Ar1 + kk + 16);
        load_frag(b0, Br0 + kk, Br0 + kk + 16);   // K = kk+half*8..+7, +16..
        load_frag(b1, Br1 + kk, Br1 + kk + 16);
        c00 = __builtin_amdgcn_wmma_f32_16x16x32_bf16(false, a0.v, false, b0.v, (short)0, c00, false, false);
        c01 = __builtin_amdgcn_wmma_f32_16x16x32_bf16(false, a0.v, false, b1.v, (short)0, c01, false, false);
        c10 = __builtin_amdgcn_wmma_f32_16x16x32_bf16(false, a1.v, false, b0.v, (short)0, c10, false, false);
        c11 = __builtin_amdgcn_wmma_f32_16x16x32_bf16(false, a1.v, false, b1.v, (short)0, c11, false, false);
    }
    // D layout: VGPR v holds row half*8+v (lanes 0-15) / rows +8 (lanes 16-31),
    // col = l16. Our indexing folds both via half.
    float* Cb = C + (size_t)(tr * 32 + half * 8) * Dc + tc * 32 + l16;
#pragma unroll
    for (int v = 0; v < 8; ++v) {
        Cb[(size_t)v * Dc]            = c00[v];
        Cb[(size_t)v * Dc + 16]       = c01[v];
        Cb[(size_t)(v + 16) * Dc]     = c10[v];
        Cb[(size_t)(v + 16) * Dc + 16]= c11[v];
    }
}

// ---------------- K4: fused row-norm + normalize + bf16 convert -------------
__global__ void k_norm_bf(const float* __restrict__ Yin, unsigned short* __restrict__ Outbf) {
    __shared__ float red[256];
    const int row = blockIdx.x;
    const float* r = Yin + (size_t)row * Dc;
    float s = 0.f;
    for (int e = threadIdx.x; e < Dc; e += 256) { float v = r[e]; s += v * v; }
    red[threadIdx.x] = s; __syncthreads();
    for (int off = 128; off > 0; off >>= 1) {
        if (threadIdx.x < off) red[threadIdx.x] += red[threadIdx.x + off];
        __syncthreads();
    }
    const float scale = 1.0f / fmaxf(sqrtf(red[0]), 1e-12f);
    for (int e = threadIdx.x; e < Dc; e += 256)
        Outbf[(size_t)row * Dc + e] = f2bf(r[e] * scale);
}

// ---------------- K5: batched similarity GEMM  S[b] = Xn[b] @ Yn[b]^T -------
// Same 2x2 register blocking; B-fragment reads Yn rows directly (they are the
// columns of Yn^T, contiguous over K).
__global__ void k_gemm_sim(const unsigned short* __restrict__ Xn,
                           const unsigned short* __restrict__ Yn,
                           float* __restrict__ S) {
    const int lane = threadIdx.x & 31;
    const int wave = threadIdx.x >> 5;
    const int b    = blockIdx.y;
    const int tile = blockIdx.x * (blockDim.x >> 5) + wave;   // 8*8 tiles/batch
    const int tr = tile >> 3, tc = tile & 7;
    const int half = lane >> 4, l16 = lane & 15;
    const unsigned short* Ar0 = Xn + (size_t)(b * Mc + tr * 32 + l16) * Dc + half * 8;
    const unsigned short* Ar1 = Ar0 + (size_t)16 * Dc;
    const unsigned short* Br0 = Yn + (size_t)(b * Mc + tc * 32 + l16) * Dc + half * 8;
    const unsigned short* Br1 = Br0 + (size_t)16 * Dc;
    v8f c00, c01, c10, c11;
#pragma unroll
    for (int v = 0; v < 8; ++v) { c00[v] = 0.f; c01[v] = 0.f; c10[v] = 0.f; c11[v] = 0.f; }
    for (int kk = 0; kk < Dc; kk += 32) {
        FragBF16 a0, a1, b0, b1;
        load_frag(a0, Ar0 + kk, Ar0 + kk + 16);
        load_frag(a1, Ar1 + kk, Ar1 + kk + 16);
        load_frag(b0, Br0 + kk, Br0 + kk + 16);
        load_frag(b1, Br1 + kk, Br1 + kk + 16);
        c00 = __builtin_amdgcn_wmma_f32_16x16x32_bf16(false, a0.v, false, b0.v, (short)0, c00, false, false);
        c01 = __builtin_amdgcn_wmma_f32_16x16x32_bf16(false, a0.v, false, b1.v, (short)0, c01, false, false);
        c10 = __builtin_amdgcn_wmma_f32_16x16x32_bf16(false, a1.v, false, b0.v, (short)0, c10, false, false);
        c11 = __builtin_amdgcn_wmma_f32_16x16x32_bf16(false, a1.v, false, b1.v, (short)0, c11, false, false);
    }
    float* Cb = S + (size_t)b * Mc * Mc + (size_t)(tr * 32 + half * 8) * Mc + tc * 32 + l16;
#pragma unroll
    for (int v = 0; v < 8; ++v) {
        Cb[(size_t)v * Mc]             = c00[v];
        Cb[(size_t)v * Mc + 16]        = c01[v];
        Cb[(size_t)(v + 16) * Mc]      = c10[v];
        Cb[(size_t)(v + 16) * Mc + 16] = c11[v];
    }
}

// ---------------- K6: per-(b,m) logsumexp CE row (one wave per row) ---------
__global__ void k_loss_rows(const float* __restrict__ S, const int* __restrict__ negs,
                            float* __restrict__ rows) {
    const int lane = threadIdx.x & 31;
    const int wave = threadIdx.x >> 5;
    const int row = blockIdx.x * (blockDim.x >> 5) + wave;    // 0..2047
    const int b = row >> 8, m = row & (Mc - 1);
    const float* Srow = S + (size_t)b * Mc * Mc + (size_t)m * Mc;
    const int* nrow = negs + (size_t)b * (NNEG * Mc) + (size_t)m * NNEG;
    float l[4];
    float mx = -INFINITY;
#pragma unroll
    for (int it = 0; it < 4; ++it) {
        const int i = lane + it * 32;
        float v = -INFINITY;
        if (i == 0) {
            v = Srow[m] * INV_TEMP;                 // positive logit = diag(S)
        } else if (i <= NNEG) {
            int j = nrow[i - 1] - b * Mc;           // local row in this batch
            v = (j == m) ? -INFINITY : Srow[j] * INV_TEMP;   // neg_is_pos mask
        }
        l[it] = v;
        mx = fmaxf(mx, v);
    }
    for (int off = 16; off > 0; off >>= 1) mx = fmaxf(mx, __shfl_xor(mx, off, 32));
    float se = 0.f;
#pragma unroll
    for (int it = 0; it < 4; ++it) se += (l[it] == -INFINITY) ? 0.f : __expf(l[it] - mx);
    for (int off = 16; off > 0; off >>= 1) se += __shfl_xor(se, off, 32);
    if (lane == 0) {
        const float lse = mx + __logf(se);
        rows[row] = lse - Srow[m] * INV_TEMP;       // lse(row) - row[0]
    }
}

// ---------------- K7: deterministic final reduction -------------------------
__global__ void k_reduce(const float* __restrict__ rows, float* __restrict__ out) {
    __shared__ float red[256];
    float s = 0.f;
    for (int i = threadIdx.x; i < ROWS; i += 256) s += rows[i];
    red[threadIdx.x] = s; __syncthreads();
    for (int off = 128; off > 0; off >>= 1) {
        if (threadIdx.x < off) red[threadIdx.x] += red[threadIdx.x + off];
        __syncthreads();
    }
    if (threadIdx.x == 0) out[0] = red[0];
}

extern "C" void kernel_launch(void* const* d_in, const int* in_sizes, int n_in,
                              void* d_out, int out_size, void* d_ws, size_t ws_size,
                              hipStream_t stream) {
    const float* cnn   = (const float*)d_in[0];
    const float* quant = (const float*)d_in[1];
    const float* Wy    = (const float*)d_in[2];
    const float* by    = (const float*)d_in[3];
    const float* Wx    = (const float*)d_in[4];
    const float* bx    = (const float*)d_in[5];
    const unsigned char* mask = (const unsigned char*)d_in[6];   // numpy bool
    const int* negs    = (const int*)d_in[7];
    (void)in_sizes; (void)n_in; (void)out_size; (void)ws_size;

    // Carve workspace (256B aligned slices), ~30 MB total.
    char* p = (char*)d_ws;
    size_t o = 0;
    auto take = [&](size_t bytes) -> char* {
        char* r = p + o;
        o = (o + bytes + 255) & ~(size_t)255;
        return r;
    };
    int*            idxbuf = (int*)take(ROWS * sizeof(int));
    unsigned short* Abf  = (unsigned short*)take((size_t)ROWS * Dc * 2);
    unsigned short* Qbf  = (unsigned short*)take((size_t)ROWS * Dc * 2);
    unsigned short* WyT  = (unsigned short*)take((size_t)Dc * Dc * 2);
    unsigned short* WxT  = (unsigned short*)take((size_t)Dc * Dc * 2);
    float*          Yp   = (float*)take((size_t)ROWS * Dc * 4);
    float*          Xp   = (float*)take((size_t)ROWS * Dc * 4);
    unsigned short* Ynbf = (unsigned short*)take((size_t)ROWS * Dc * 2);
    unsigned short* Xnbf = (unsigned short*)take((size_t)ROWS * Dc * 2);
    float*          Smat = (float*)take((size_t)Bc * Mc * Mc * 4);
    float*          lrow = (float*)take((size_t)ROWS * 4);

    k_wt<<<(Dc * Dc + 255) / 256, 256, 0, stream>>>(Wy, WyT);
    k_wt<<<(Dc * Dc + 255) / 256, 256, 0, stream>>>(Wx, WxT);
    k_mask_idx<<<1, Bc, 0, stream>>>(mask, idxbuf);
    k_gather<<<ROWS, 256, 0, stream>>>(cnn, quant, idxbuf, Abf, Qbf);
    // 64 row-tiles x 24 col-tiles = 1536 wave-tiles, 8 waves/block -> 192 blocks
    k_gemm_proj<<<192, 256, 0, stream>>>(Abf, WyT, by, Yp);
    k_gemm_proj<<<192, 256, 0, stream>>>(Qbf, WxT, bx, Xp);
    k_norm_bf<<<ROWS, 256, 0, stream>>>(Yp, Ynbf);
    k_norm_bf<<<ROWS, 256, 0, stream>>>(Xp, Xnbf);
    // 64 wave-tiles per batch, 8 waves/block -> (8, 8) blocks
    k_gemm_sim<<<dim3(8, Bc), 256, 0, stream>>>(Xnbf, Ynbf, Smat);
    k_loss_rows<<<ROWS / 8, 256, 0, stream>>>(Smat, negs, lrow);
    k_reduce<<<1, 256, 0, stream>>>(lrow, (float*)d_out);
}
